// AoAReader_35424890257620
// MI455X (gfx1250) — compile-verified, matrix operand order checked by reading, not run on hardware
//
#include <hip/hip_runtime.h>

// ---------------- problem constants ----------------
constexpr int Bb = 32, Dd = 1000, Qq = 50, Vv = 50000, Ee = 384, Hh = 384, Cc = 10;
constexpr int G3 = 3 * Hh;   // 1152 gates
constexpr int H2 = 2 * Hh;   // 768 bi-directional hidden
constexpr int QP = 64;       // Q padded to 64 for tiling
constexpr int MT_D = 63;     // ceil(1000/16)
constexpr int MT_Q = 4;      // ceil(50/16)
constexpr int NT_G = 72;     // 1152/16
constexpr int NTILE = 6;     // N-tiles per job (A-fragment reuse)
constexpr int NGRP = NT_G / NTILE;  // 12 groups

typedef unsigned short u16;
typedef __attribute__((ext_vector_type(16))) __bf16 v16bf;
typedef __attribute__((ext_vector_type(8)))  float  v8f;

// ---------------- bf16 helpers ----------------
__device__ __forceinline__ u16 f2bf(float f) {
  unsigned u = __builtin_bit_cast(unsigned, f);
  u += 0x7FFFu + ((u >> 16) & 1u);           // round-to-nearest-even
  return (u16)(u >> 16);
}
__device__ __forceinline__ float bf2f(u16 h) {
  unsigned u = ((unsigned)h) << 16;
  return __builtin_bit_cast(float, u);
}
// load 8 consecutive bf16 (16B) into fragment elements [off..off+7]
__device__ __forceinline__ void ld8(v16bf& v, int off, const u16* p) {
  uint4 d = *(const uint4*)p;
  unsigned w0 = d.x, w1 = d.y, w2 = d.z, w3 = d.w;
  v[off + 0] = __builtin_bit_cast(__bf16, (u16)(w0 & 0xFFFFu));
  v[off + 1] = __builtin_bit_cast(__bf16, (u16)(w0 >> 16));
  v[off + 2] = __builtin_bit_cast(__bf16, (u16)(w1 & 0xFFFFu));
  v[off + 3] = __builtin_bit_cast(__bf16, (u16)(w1 >> 16));
  v[off + 4] = __builtin_bit_cast(__bf16, (u16)(w2 & 0xFFFFu));
  v[off + 5] = __builtin_bit_cast(__bf16, (u16)(w2 >> 16));
  v[off + 6] = __builtin_bit_cast(__bf16, (u16)(w3 & 0xFFFFu));
  v[off + 7] = __builtin_bit_cast(__bf16, (u16)(w3 >> 16));
}
__device__ __forceinline__ v8f wmma_bf16(v16bf a, v16bf b, v8f c) {
  return __builtin_amdgcn_wmma_f32_16x16x32_bf16(false, a, false, b, (short)0, c,
                                                 false, false);
}
__device__ __forceinline__ float sigm(float x) { return 1.f / (1.f + __expf(-x)); }

// ---------------- kernel: f32 -> bf16 conversions ----------------
__global__ void k_convert(const float* __restrict__ emb, const float* __restrict__ wihf,
                          const float* __restrict__ whhf, const float* __restrict__ wihb,
                          const float* __restrict__ whhb, u16* __restrict__ emb_bf,
                          u16* __restrict__ wihf_bf, u16* __restrict__ whhf_bf,
                          u16* __restrict__ wihb_bf, u16* __restrict__ whhb_bf) {
  int i = blockIdx.x * 256 + threadIdx.x;
  const int NE = Vv * Ee, NW = G3 * Ee;
  if (i < NE) { emb_bf[i] = f2bf(emb[i]); return; }
  i -= NE;
  if (i < NW) { wihf_bf[i] = f2bf(wihf[i]); return; }
  i -= NW;
  if (i < NW) { whhf_bf[i] = f2bf(whhf[i]); return; }
  i -= NW;
  if (i < NW) { wihb_bf[i] = f2bf(wihb[i]); return; }
  i -= NW;
  if (i < NW) { whhb_bf[i] = f2bf(whhb[i]); return; }
}

// ---------------- kernel: fused gather + input projection GEMM (WMMA) -------
// xw[b,t,g] = emb[ids[b, rid(t)]] . Wih[g,:] + bih[g]   (rid folds bwd reversal)
// One job = 16 t-rows x 6 gate-tiles; A fragment loaded once per K-step and
// reused across the 6 N-tiles (6 accumulators). K-loop unrolled 2x so next
// iteration's loads issue under this iteration's WMMAs.
__global__ void __launch_bounds__(256) k_xw(
    const int* __restrict__ doc_ids, const int* __restrict__ doc_len,
    const int* __restrict__ qry_ids, const int* __restrict__ qry_len,
    const u16* __restrict__ emb_bf, const u16* __restrict__ wih_f,
    const u16* __restrict__ wih_b, const float* __restrict__ bih_f,
    const float* __restrict__ bih_b, u16* __restrict__ xw_doc,
    u16* __restrict__ xw_qry) {
  const int lane = threadIdx.x & 31;
  const int wv = threadIdx.x >> 5;
  int job = blockIdx.x * 8 + wv;
  const int DOCJ = 2 * Bb * MT_D * NGRP;
  const int QRYJ = 2 * Bb * MT_Q * NGRP;
  int seq, dir, b, mt, grp;
  if (job < DOCJ) {
    seq = 0;
    grp = job % NGRP; job /= NGRP;
    mt = job % MT_D; job /= MT_D;
    b = job % Bb; dir = job / Bb;
  } else {
    job -= DOCJ;
    if (job >= QRYJ) return;
    seq = 1;
    grp = job % NGRP; job /= NGRP;
    mt = job % MT_Q; job /= MT_Q;
    b = job % Bb; dir = job / Bb;
  }
  const int T = seq ? Qq : Dd;
  const int* ids = (seq ? qry_ids : doc_ids) + b * T;
  const int len = (seq ? qry_len : doc_len)[b];
  const u16* wih = dir ? wih_b : wih_f;
  const float* bih = dir ? bih_b : bih_f;
  u16* xw = seq ? (xw_qry + (size_t)dir * Bb * Qq * G3)
                : (xw_doc + (size_t)dir * Bb * Dd * G3);

  const int halfk = lane >> 4;
  const int lrow = lane & 15;
  int t = mt * 16 + lrow;
  int tc = t < T ? t : T - 1;
  int rid = dir ? (tc < len ? len - 1 - tc : tc) : tc;
  const u16* arow = emb_bf + (size_t)ids[rid] * Ee;

  const u16* brow[NTILE];
#pragma unroll
  for (int jj = 0; jj < NTILE; ++jj)
    brow[jj] = wih + (size_t)((grp * NTILE + jj) * 16 + lrow) * Ee;

  v8f acc[NTILE];
#pragma unroll
  for (int jj = 0; jj < NTILE; ++jj) acc[jj] = (v8f){};

#pragma unroll 2
  for (int kk = 0; kk < Ee / 32; ++kk) {
    int k0 = kk * 32 + halfk * 8;
    v16bf a;
    ld8(a, 0, arow + k0);
    ld8(a, 8, arow + k0 + 16);
#pragma unroll
    for (int jj = 0; jj < NTILE; ++jj) {
      v16bf bm;
      ld8(bm, 0, brow[jj] + k0);
      ld8(bm, 8, brow[jj] + k0 + 16);
      acc[jj] = wmma_bf16(a, bm, acc[jj]);
    }
  }
  u16* out_b = xw + (size_t)b * T * G3;
#pragma unroll
  for (int jj = 0; jj < NTILE; ++jj) {
    int g = (grp * NTILE + jj) * 16 + lrow;
    float bias = bih[g];
#pragma unroll
    for (int i = 0; i < 8; ++i) {
      int to = mt * 16 + halfk * 8 + i;
      if (to < T) out_b[(size_t)to * G3 + g] = f2bf(acc[jj][i] + bias);
    }
  }
}

// ---------------- kernel: sequential GRU scan (WMMA per step) ---------------
// grid.x = 8: bit0 = batch-half, bit1 = direction, bit2 = seq (0=doc,1=qry)
__global__ void __launch_bounds__(384) k_scan(
    const int* __restrict__ doc_len, const int* __restrict__ qry_len,
    const u16* __restrict__ whh_f_bf, const u16* __restrict__ whh_b_bf,
    const float* __restrict__ bhh_f, const float* __restrict__ bhh_b,
    const u16* __restrict__ xw_doc, const u16* __restrict__ xw_qry,
    u16* __restrict__ dos, u16* __restrict__ qos) {
  const int half_b = blockIdx.x & 1;
  const int dir = (blockIdx.x >> 1) & 1;
  const int seq = blockIdx.x >> 2;
  const int T = seq ? Qq : Dd;
  const int* lens = seq ? qry_len : doc_len;
  const u16* whh = dir ? whh_b_bf : whh_f_bf;
  const float* bhh = dir ? bhh_b : bhh_f;
  const u16* xw = seq ? (xw_qry + (size_t)dir * Bb * Qq * G3)
                      : (xw_doc + (size_t)dir * Bb * Dd * G3);
  u16* out = seq ? qos : dos;

  extern __shared__ char smem[];
  const int HP = 392;  // padded bf16 row (16B-aligned rows, bank-spread)
  float* h_f = (float*)smem;                               // [16][384]
  u16* h_bf = (u16*)(smem + 16 * 384 * 4);                 // [16][392]
  float* hw = (float*)(smem + 16 * 384 * 4 + 16 * HP * 2); // [16][1152]

  const int tid = threadIdx.x;
  const int lane = tid & 31, wv = tid >> 5;
  const int lrow = lane & 15, halfk = lane >> 4;

  for (int i = tid; i < 16 * 384; i += 384) h_f[i] = 0.f;
  for (int i = tid; i < 16 * HP; i += 384) h_bf[i] = 0;

  // ---- step-invariant hoists ----
  // phase-1 B row pointers (6 N-tiles per wave)
  const u16* browj[6];
#pragma unroll
  for (int j6 = 0; j6 < 6; ++j6)
    browj[j6] = whh + (size_t)((wv + 12 * j6) * 16 + lrow) * Hh;
  const u16* arow = h_bf + lrow * HP;
  // phase-2 invariants: biases, lengths, row bases (constant-indexed via full unroll)
  float bh0 = bhh[tid], bh1 = bhh[Hh + tid], bh2 = bhh[2 * Hh + tid];
  int len_r[16];
  int row0[16];
#pragma unroll
  for (int bb = 0; bb < 16; ++bb) {
    int bg = half_b * 16 + bb;
    len_r[bb] = lens[bg];
    row0[bb] = bg * T;
  }
  // prefetch lane mapping: 16 rows x 18 cachelines (128B) covers 16x2304B slab
  const int pf_bb = tid / 18, pf_c = tid % 18;
  const int pf_row0 = (half_b * 16 + pf_bb) * T;

  __syncthreads();

  for (int t = 0; t < T; ++t) {
    // prefetch next step's xw slab while the GEMM runs (global_prefetch_b8)
    if (t + 1 < T && tid < 288)
      __builtin_prefetch(xw + (size_t)(pf_row0 + t + 1) * G3 + pf_c * 64, 0, 0);

    // phase 1: hw = h @ Whh^T  -- A fragment loaded once per K-step,
    // reused across 6 N-tiles (6 accumulators); K-loop unrolled 2x
    v8f acc[6];
#pragma unroll
    for (int j6 = 0; j6 < 6; ++j6) acc[j6] = (v8f){};
#pragma unroll 2
    for (int kk = 0; kk < Hh / 32; ++kk) {
      int k0 = kk * 32 + halfk * 8;
      v16bf a;
      ld8(a, 0, arow + k0);
      ld8(a, 8, arow + k0 + 16);
#pragma unroll
      for (int j6 = 0; j6 < 6; ++j6) {
        v16bf bm;
        ld8(bm, 0, browj[j6] + k0);
        ld8(bm, 8, browj[j6] + k0 + 16);
        acc[j6] = wmma_bf16(a, bm, acc[j6]);
      }
    }
#pragma unroll
    for (int j6 = 0; j6 < 6; ++j6) {
      int gg = (wv + 12 * j6) * 16 + lrow;
#pragma unroll
      for (int i = 0; i < 8; ++i) hw[(halfk * 8 + i) * G3 + gg] = acc[j6][i];
    }
    __syncthreads();

    // phase 2: gates + state update; thread tid owns hidden index j=tid
    {
      const int j = tid;
#pragma unroll
      for (int bb = 0; bb < 16; ++bb) {
        int len = len_r[bb];
        bool valid = t < len;
        const u16* xrow = xw + (size_t)(row0[bb] + t) * G3;
        float xr = bf2f(xrow[j]);
        float xz = bf2f(xrow[Hh + j]);
        float xn = bf2f(xrow[2 * Hh + j]);
        float hr = hw[bb * G3 + j] + bh0;
        float hz = hw[bb * G3 + Hh + j] + bh1;
        float hn = hw[bb * G3 + 2 * Hh + j] + bh2;
        float r = sigm(xr + hr);
        float z = sigm(xz + hz);
        float n = tanhf(xn + r * hn);
        float hold = h_f[bb * 384 + j];
        float hnew = (1.f - z) * n + z * hold;
        float hsel = valid ? hnew : hold;
        h_f[bb * 384 + j] = hsel;
        h_bf[bb * HP + j] = f2bf(hsel);
        int orid = dir ? (t < len ? len - 1 - t : t) : t;  // un-reverse on store
        out[(size_t)(row0[bb] + orid) * H2 + dir * Hh + j] =
            f2bf(valid ? hnew : 0.f);
      }
    }
    __syncthreads();
  }
}

// ---------------- kernel: M = dos . qos^T (WMMA) ----------------
// One job = 16 d-rows x all 4 q-tiles; A reused across tiles.
__global__ void __launch_bounds__(256) k_scores(const u16* __restrict__ dos,
                                                const u16* __restrict__ qos,
                                                float* __restrict__ Mbuf) {
  const int lane = threadIdx.x & 31, wv = threadIdx.x >> 5;
  int job = blockIdx.x * 8 + wv;
  if (job >= Bb * MT_D) return;
  int mt = job % MT_D;
  int b = job / MT_D;
  const int lrow = lane & 15, halfk = lane >> 4;
  int d = mt * 16 + lrow; if (d > Dd - 1) d = Dd - 1;
  const u16* arow = dos + ((size_t)b * Dd + d) * H2;
  const u16* brow[MT_Q];
#pragma unroll
  for (int nt = 0; nt < MT_Q; ++nt) {
    int q = nt * 16 + lrow; if (q > Qq - 1) q = Qq - 1;
    brow[nt] = qos + ((size_t)b * Qq + q) * H2;
  }
  v8f acc[MT_Q];
#pragma unroll
  for (int nt = 0; nt < MT_Q; ++nt) acc[nt] = (v8f){};
#pragma unroll 2
  for (int kk = 0; kk < H2 / 32; ++kk) {
    int k0 = kk * 32 + halfk * 8;
    v16bf a;
    ld8(a, 0, arow + k0);
    ld8(a, 8, arow + k0 + 16);
#pragma unroll
    for (int nt = 0; nt < MT_Q; ++nt) {
      v16bf bm;
      ld8(bm, 0, brow[nt] + k0);
      ld8(bm, 8, brow[nt] + k0 + 16);
      acc[nt] = wmma_bf16(a, bm, acc[nt]);
    }
  }
  float* mrow = Mbuf + (size_t)b * Dd * QP;
#pragma unroll
  for (int nt = 0; nt < MT_Q; ++nt) {
#pragma unroll
    for (int i = 0; i < 8; ++i) {
      int dd = mt * 16 + halfk * 8 + i;
      int qq = nt * 16 + lrow;
      if (dd < Dd) mrow[(size_t)dd * QP + qq] = acc[nt][i];
    }
  }
}

// ---------------- small reduction kernels ----------------
__global__ void k_zero(float* __restrict__ p, int n) {
  int i = blockIdx.x * 256 + threadIdx.x;
  if (i < n) p[i] = 0.f;
}

// beta row-softmax over q, accumulate into avg_beta
__global__ void k_beta(const float* __restrict__ Mbuf, const int* __restrict__ dlen_p,
                       const int* __restrict__ qlen_p, float* __restrict__ avg_beta) {
  int b = blockIdx.x / Dd, d = blockIdx.x % Dd;
  if (d >= dlen_p[b]) return;
  int ql = qlen_p[b];
  __shared__ float red[64];
  int tid = threadIdx.x;
  const float* row = Mbuf + ((size_t)b * Dd + d) * QP;
  float x = (tid < Qq) ? row[tid] : -3.4e38f;     // unmasked max, matches ref
  red[tid] = x; __syncthreads();
  for (int s = 32; s > 0; s >>= 1) {
    if (tid < s) red[tid] = fmaxf(red[tid], red[tid + s]);
    __syncthreads();
  }
  float mx = red[0]; __syncthreads();
  float e = (tid < Qq && tid < ql) ? __expf(x - mx) : 0.f;
  red[tid] = e; __syncthreads();
  for (int s = 32; s > 0; s >>= 1) {
    if (tid < s) red[tid] += red[tid + s];
    __syncthreads();
  }
  float sum = red[0];
  if (tid < Qq) atomicAdd(&avg_beta[b * QP + tid], e / (sum + 1e-12f));
}

// per-(b,q): colmax/colsum for alpha; also avg_beta /= docs_len
__global__ void k_colstats(const float* __restrict__ Mbuf,
                           const int* __restrict__ dlen_p,
                           const int* __restrict__ qlen_p, float* __restrict__ colmax,
                           float* __restrict__ colsum, float* __restrict__ avg_beta) {
  int b = blockIdx.x / Qq, q = blockIdx.x % Qq;
  int tid = threadIdx.x;
  int dl = dlen_p[b], ql = qlen_p[b];
  __shared__ float red[256];
  const float* mb = Mbuf + (size_t)b * Dd * QP + q;
  float mx = -3.4e38f;
  for (int d = tid; d < Dd; d += 256) mx = fmaxf(mx, mb[(size_t)d * QP]);
  red[tid] = mx; __syncthreads();
  for (int s = 128; s > 0; s >>= 1) {
    if (tid < s) red[tid] = fmaxf(red[tid], red[tid + s]);
    __syncthreads();
  }
  mx = red[0]; __syncthreads();
  float sm = 0.f;
  bool qv = q < ql;
  for (int d = tid; d < Dd; d += 256) {
    float e = __expf(mb[(size_t)d * QP] - mx);
    if (d < dl && qv) sm += e;
  }
  red[tid] = sm; __syncthreads();
  for (int s = 128; s > 0; s >>= 1) {
    if (tid < s) red[tid] += red[tid + s];
    __syncthreads();
  }
  if (tid == 0) {
    colmax[b * QP + q] = mx;
    colsum[b * QP + q] = red[0];
    avg_beta[b * QP + q] = avg_beta[b * QP + q] / (float)dl;
  }
}

// s[b,d] = sum_q alpha[b,d,q] * avg_beta[b,q]
__global__ void k_s(const float* __restrict__ Mbuf, const float* __restrict__ colmax,
                    const float* __restrict__ colsum,
                    const float* __restrict__ avg_beta,
                    const int* __restrict__ dlen_p, const int* __restrict__ qlen_p,
                    float* __restrict__ sbuf) {
  int i = blockIdx.x * 256 + threadIdx.x;
  if (i >= Bb * Dd) return;
  int b = i / Dd, d = i % Dd;
  int dl = dlen_p[b], ql = qlen_p[b];
  float acc = 0.f;
  if (d < dl) {
    const float* row = Mbuf + ((size_t)b * Dd + d) * QP;
    for (int q = 0; q < Qq; ++q) {
      if (q < ql) {
        float alpha = __expf(row[q] - colmax[b * QP + q]) / (colsum[b * QP + q] + 1e-12f);
        acc += alpha * avg_beta[b * QP + q];
      }
    }
  }
  sbuf[i] = acc;
}

// pb[b,c] = sum_d [doc[b,d]==cand[b,c]] * s[b,d]
__global__ void k_pb(const int* __restrict__ doc_ids, const int* __restrict__ cands,
                     const float* __restrict__ sbuf, float* __restrict__ out) {
  int b = blockIdx.x / Cc, c = blockIdx.x % Cc;
  int tid = threadIdx.x;
  int cand = cands[b * Cc + c];
  __shared__ float red[256];
  float acc = 0.f;
  for (int d = tid; d < Dd; d += 256)
    if (doc_ids[b * Dd + d] == cand) acc += sbuf[b * Dd + d];
  red[tid] = acc; __syncthreads();
  for (int s = 128; s > 0; s >>= 1) {
    if (tid < s) red[tid] += red[tid + s];
    __syncthreads();
  }
  if (tid == 0) out[b * Cc + c] = red[0];
}

// ---------------- launcher ----------------
extern "C" void kernel_launch(void* const* d_in, const int* in_sizes, int n_in,
                              void* d_out, int out_size, void* d_ws, size_t ws_size,
                              hipStream_t stream) {
  const int* docs_input = (const int*)d_in[0];
  const int* docs_len   = (const int*)d_in[1];
  const int* querys_input = (const int*)d_in[2];
  const int* querys_len   = (const int*)d_in[3];
  const int* candidates   = (const int*)d_in[4];
  const float* emb   = (const float*)d_in[5];
  const float* Wih_f = (const float*)d_in[6];
  const float* Whh_f = (const float*)d_in[7];
  const float* bih_f = (const float*)d_in[8];
  const float* bhh_f = (const float*)d_in[9];
  const float* Wih_b = (const float*)d_in[10];
  const float* Whh_b = (const float*)d_in[11];
  const float* bih_b = (const float*)d_in[12];
  const float* bhh_b = (const float*)d_in[13];
  float* out = (float*)d_out;

  // workspace carve (all 256B aligned)
  char* ws = (char*)d_ws;
  size_t off = 0;
  auto take = [&](size_t bytes) { char* p = ws + off; off += (bytes + 255) & ~(size_t)255; return p; };
  u16* emb_bf  = (u16*)take((size_t)Vv * Ee * 2);
  u16* wihf_bf = (u16*)take((size_t)G3 * Ee * 2);
  u16* whhf_bf = (u16*)take((size_t)G3 * Ee * 2);
  u16* wihb_bf = (u16*)take((size_t)G3 * Ee * 2);
  u16* whhb_bf = (u16*)take((size_t)G3 * Ee * 2);
  u16* xw_doc  = (u16*)take((size_t)2 * Bb * Dd * G3 * 2);
  u16* xw_qry  = (u16*)take((size_t)2 * Bb * Qq * G3 * 2);
  u16* dos     = (u16*)take((size_t)Bb * Dd * H2 * 2);
  u16* qos     = (u16*)take((size_t)Bb * Qq * H2 * 2);
  float* Mbuf  = (float*)take((size_t)Bb * Dd * QP * 4);
  float* avg_beta = (float*)take((size_t)Bb * QP * 4);
  float* colmax   = (float*)take((size_t)Bb * QP * 4);
  float* colsum   = (float*)take((size_t)Bb * QP * 4);
  float* sbuf     = (float*)take((size_t)Bb * Dd * 4);
  (void)ws_size; (void)n_in; (void)in_sizes; (void)out_size;

  // 1) bf16 conversions
  {
    int total = Vv * Ee + 4 * G3 * Ee;
    k_convert<<<(total + 255) / 256, 256, 0, stream>>>(
        emb, Wih_f, Whh_f, Wih_b, Whh_b, emb_bf, wihf_bf, whhf_bf, wihb_bf, whhb_bf);
  }
  // 2) gather + input-projection GEMM (doc fwd/bwd, qry fwd/bwd)
  {
    int jobs = 2 * Bb * MT_D * NGRP + 2 * Bb * MT_Q * NGRP;  // 51456
    k_xw<<<jobs / 8, 256, 0, stream>>>(docs_input, docs_len, querys_input, querys_len,
                                       emb_bf, wihf_bf, wihb_bf, bih_f, bih_b,
                                       xw_doc, xw_qry);
  }
  // 3) GRU scans: 8 blocks (batch-half x dir x seq), 110848B dynamic LDS
  {
    size_t lds = 16 * 384 * 4 + 16 * 392 * 2 + 16 * 1152 * 4;
    k_scan<<<8, 384, lds, stream>>>(docs_len, querys_len, whhf_bf, whhb_bf,
                                    bhh_f, bhh_b, xw_doc, xw_qry, dos, qos);
  }
  // 4) score matrix M
  k_scores<<<(Bb * MT_D + 7) / 8, 256, 0, stream>>>(dos, qos, Mbuf);
  // 5) attention-over-attention
  k_zero<<<(Bb * QP + 255) / 256, 256, 0, stream>>>(avg_beta, Bb * QP);
  k_beta<<<Bb * Dd, 64, 0, stream>>>(Mbuf, docs_len, querys_len, avg_beta);
  k_colstats<<<Bb * Qq, 256, 0, stream>>>(Mbuf, docs_len, querys_len, colmax, colsum,
                                          avg_beta);
  k_s<<<(Bb * Dd + 255) / 256, 256, 0, stream>>>(Mbuf, colmax, colsum, avg_beta,
                                                 docs_len, querys_len, sbuf);
  // 6) candidate pointer sums
  k_pb<<<Bb * Cc, 256, 0, stream>>>(docs_input, candidates, sbuf, out);
}
